// LearningModel_16922171146243
// MI455X (gfx1250) — compile-verified, matrix operand order checked by reading, not run on hardware
//
#include <hip/hip_runtime.h>
#include <hip/hip_bf16.h>
#include <math.h>

// ---------------------------------------------------------------------------
// Problem constants (from the reference)
// ---------------------------------------------------------------------------
#define Fdim 64
#define Hdim 512
#define Ddim 256
#define TPM_C 0.1f
#define EC_C  0.01f

typedef __bf16 v16bf __attribute__((ext_vector_type(16)));
typedef float  v8f   __attribute__((ext_vector_type(8)));

__device__ __forceinline__ unsigned short f32_to_bf16_bits(float f) {
    unsigned int u = __float_as_uint(f);
    unsigned int r = u + 0x7FFFu + ((u >> 16) & 1u);   // round-to-nearest-even
    return (unsigned short)(r >> 16);
}

// Fragment element layout for 16-bit WMMA operands (A 16x32 / B 32x16):
//   lane half = lane>>4, element i -> K%32 = (i<8 ? half*8+i : 16+half*8+(i-8))
// Inverse (used when swizzling): kk=K%32, seg=kk>>3:
//   half = seg&1 ; i = (seg>>1)*8 + (kk&7)

// ---------------------------------------------------------------------------
// Kernel 1: setup.  fm/fa (64), v = W2 @ wvec (512), c0 = b2 . wvec,
// W1 -> bf16 pre-swizzled into per-lane-contiguous fragment order, zero out.
// W1sw layout: ((htile*2 + kb)*32 + lane)*16 + i   (htile<32, kb<2, lane<32)
// ---------------------------------------------------------------------------
__global__ __launch_bounds__(256)
void setup_kernel(const float* __restrict__ tweak,
                  const float* __restrict__ mulW,   // (64,8)
                  const float* __restrict__ mulb,   // (64,)
                  const float* __restrict__ W1,     // (64,512)
                  const float* __restrict__ W2,     // (512,256)
                  const float* __restrict__ b2,     // (256,)
                  const float* __restrict__ key_w,  // (256,)
                  unsigned short* __restrict__ W1sw,
                  float* __restrict__ fm, float* __restrict__ fa,
                  float* __restrict__ v,  float* __restrict__ c0,
                  float* __restrict__ out, int S_) {
    __shared__ float swv[Ddim];
    const int tid = threadIdx.x;

    if (tid < Fdim) {
        float m = 0.f, a = 0.f;
        #pragma unroll
        for (int g = 0; g < 8; ++g) {
            float w = mulW[tid * 8 + g];
            m += w * tweak[g];
            a += w * tweak[8 + g];
        }
        fm[tid] = m + mulb[tid];
        fa[tid] = a + mulb[tid];
    }
    // wvec[d] = key_w[d] * final_blend[d % 8]
    swv[tid] = key_w[tid] * tweak[16 + (tid & 7)];
    __syncthreads();

    // v[h] = sum_d W2[h,d] * wvec[d]
    for (int h = tid; h < Hdim; h += 256) {
        const float* row = W2 + (size_t)h * Ddim;
        float s = 0.f;
        for (int d = 0; d < Ddim; ++d) s += row[d] * swv[d];
        v[h] = s;
    }
    if (tid == 0) {
        float s = 0.f;
        for (int d = 0; d < Ddim; ++d) s += b2[d] * swv[d];
        *c0 = s;
    }

    // Swizzled bf16 conversion of W1: dest idx -> source (K, hcol)
    for (int idx = tid; idx < Fdim * Hdim; idx += 256) {
        const int i    = idx & 15;
        const int lane = (idx >> 4) & 31;
        const int f    = idx >> 9;          // htile*2 + kb
        const int htile = f >> 1;
        const int kb    = f & 1;
        const int half  = lane >> 4;
        const int nn    = lane & 15;
        const int koff  = half * 8;
        const int K = kb * 32 + ((i < 8) ? (koff + i) : (16 + koff + (i - 8)));
        W1sw[idx] = f32_to_bf16_bits(W1[K * Hdim + htile * 16 + nn]);
    }

    if (tid < 5) out[tid] = 0.f;
    if (tid == 5) out[5] = (float)S_;
}

// ---------------------------------------------------------------------------
// Kernel 2: exclusive scan of seg_sizes -> offsets  (single block)
// ---------------------------------------------------------------------------
__global__ __launch_bounds__(256)
void scan_kernel(const int* __restrict__ seg_sizes, int* __restrict__ offsets, int S_) {
    __shared__ int part[256];
    __shared__ int base[256];
    const int tid = threadIdx.x;
    const int chunk = (S_ + 255) / 256;
    const int lo = tid * chunk;
    const int hi = (lo + chunk < S_) ? (lo + chunk) : S_;
    int s = 0;
    for (int i = lo; i < hi; ++i) s += seg_sizes[i];
    part[tid] = s;
    __syncthreads();
    if (tid == 0) {
        int run = 0;
        for (int i = 0; i < 256; ++i) { base[i] = run; run += part[i]; }
    }
    __syncthreads();
    int run = base[tid];
    for (int i = lo; i < hi; ++i) { offsets[i] = run; run += seg_sizes[i]; }
}

// ---------------------------------------------------------------------------
// Kernel 3: fused  fv -> relu(fv@W1+b1) . v + c0  ->  logits   (WMMA bf16)
// 32 rows per block (2 row-tiles), 8 waves, wave w owns H-tiles [4w, 4w+4).
// A fragments staged swizzled in LDS (2 x ds_load_b128 per fragment),
// B fragments read pre-swizzled from global (2 x global_load_b128).
// ---------------------------------------------------------------------------
__global__ __launch_bounds__(256)
void fused_mlp_logits_kernel(const float* __restrict__ features,
                             const unsigned short* __restrict__ W1sw,
                             const float* __restrict__ b1,            // (512,)
                             const float* __restrict__ fm,
                             const float* __restrict__ fa,
                             const float* __restrict__ v,             // (512,)
                             const float* __restrict__ c0,
                             float* __restrict__ logits, int N_) {
    // fv tile in fragment order: ((rt*2+kb)*32 + lane)*16 + i   (4KB)
    __shared__ alignas(16) unsigned short sfvsw[2 * 2 * 32 * 16];
    __shared__ float slog[32];

    const int tid  = threadIdx.x;
    const int lane = tid & 31;
    const int wave = tid >> 5;
    const int rowbase = blockIdx.x * 32;

    if (tid < 32) slog[tid] = 0.f;

    // Stage fv (32x64 = 2048 elements, 8 per thread) directly at swizzled pos.
    {
        const int i0 = tid * 8;
        const int r  = i0 >> 6;            // 0..31
        const int k0 = i0 & 63;            // multiple of 8
        int grow = rowbase + r;
        if (grow > N_ - 1) grow = N_ - 1;  // uniform-safe clamp
        const float* frow = features + (size_t)grow * Fdim;
        const int rt = r >> 4;
        const int rr = r & 15;
        #pragma unroll
        for (int j = 0; j < 8; ++j) {
            const int k  = k0 + j;
            const int kb = k >> 5;
            const int kk = k & 31;
            const int seg  = kk >> 3;
            const int half = seg & 1;
            const int ii   = (seg >> 1) * 8 + (kk & 7);
            const int dlane = half * 16 + rr;
            sfvsw[((rt * 2 + kb) * 32 + dlane) * 16 + ii] =
                f32_to_bf16_bits(fm[k] * frow[k] + fa[k]);
        }
    }
    __syncthreads();

    const int half = lane >> 4;
    const int n    = lane & 15;

    union Frag { v16bf v; uint4 q[2]; };

    // A fragments: [rt][kb], each two ds_load_b128
    Frag afrag[2][2];
    #pragma unroll
    for (int rt = 0; rt < 2; ++rt) {
        #pragma unroll
        for (int kb = 0; kb < 2; ++kb) {
            const uint4* p =
                (const uint4*)&sfvsw[((rt * 2 + kb) * 32 + lane) * 16];
            afrag[rt][kb].q[0] = p[0];
            afrag[rt][kb].q[1] = p[1];
        }
    }

    float racc[2][8];
    #pragma unroll
    for (int rt = 0; rt < 2; ++rt)
        #pragma unroll
        for (int j = 0; j < 8; ++j) racc[rt][j] = 0.f;

    #pragma unroll
    for (int t = 0; t < 4; ++t) {
        const int htile = wave * 4 + t;
        const int hcol  = htile * 16 + n;            // this lane's H column
        v8f acc0 = {0.f,0.f,0.f,0.f,0.f,0.f,0.f,0.f};
        v8f acc1 = {0.f,0.f,0.f,0.f,0.f,0.f,0.f,0.f};
        #pragma unroll
        for (int kb = 0; kb < 2; ++kb) {
            Frag bfrag;
            const uint4* p =
                (const uint4*)&W1sw[((htile * 2 + kb) * 32 + lane) * 16];
            bfrag.q[0] = p[0];
            bfrag.q[1] = p[1];
            acc0 = __builtin_amdgcn_wmma_f32_16x16x32_bf16(
                       false, afrag[0][kb].v, false, bfrag.v,
                       (short)0, acc0, false, false);
            acc1 = __builtin_amdgcn_wmma_f32_16x16x32_bf16(
                       false, afrag[1][kb].v, false, bfrag.v,
                       (short)0, acc1, false, false);
        }
        const float bias = b1[hcol];
        const float vh   = v[hcol];
        #pragma unroll
        for (int j = 0; j < 8; ++j) {                // C: VGPR j -> row j / j+8
            float x0 = fmaxf(acc0[j] + bias, 0.f);
            float x1 = fmaxf(acc1[j] + bias, 0.f);
            racc[0][j] += x0 * vh;
            racc[1][j] += x1 * vh;
        }
    }

    // Reduce over the 16 columns of each half-wave (xor masks stay within half)
    #pragma unroll
    for (int rt = 0; rt < 2; ++rt)
        #pragma unroll
        for (int j = 0; j < 8; ++j)
            #pragma unroll
            for (int m = 1; m < 16; m <<= 1)
                racc[rt][j] += __shfl_xor(racc[rt][j], m, 32);

    if (n == 0) {                                    // lanes 0 and 16
        #pragma unroll
        for (int rt = 0; rt < 2; ++rt)
            #pragma unroll
            for (int j = 0; j < 8; ++j)
                atomicAdd(&slog[rt * 16 + half * 8 + j], racc[rt][j]);
    }
    __syncthreads();

    if (tid < 32) {
        const int row = rowbase + tid;
        if (row < N_) logits[row] = slog[tid] + c0[0];
    }
}

// ---------------------------------------------------------------------------
// Kernel 4: per-segment log-softmax losses. One block per segment, 2 passes.
// All loss terms in closed form from: m, S1=sum e^x, S2=sum x e^x,
// G1=sum g*x, G2=sum g   (x = logit - m).
// ---------------------------------------------------------------------------
__global__ __launch_bounds__(256)
void segment_loss_kernel(const float* __restrict__ logits,
                         const int* __restrict__ flat_idx,
                         const float* __restrict__ good_mask,
                         const float* __restrict__ event_time,
                         const int* __restrict__ sel_flat_pos,
                         const int* __restrict__ seg_sizes,
                         const int* __restrict__ offsets,
                         float* __restrict__ out) {
    __shared__ float redA[256];
    __shared__ float redB[256];
    __shared__ float sm;
    __shared__ float sS1, sS2;

    const int b    = blockIdx.x;
    const int tid  = threadIdx.x;
    const int off  = offsets[b];
    const int size = seg_sizes[b];

    // pass 1: segment max
    float lmax = -INFINITY;
    for (int i = tid; i < size; i += 256)
        lmax = fmaxf(lmax, logits[flat_idx[off + i]]);
    redA[tid] = lmax;
    __syncthreads();
    for (int s = 128; s > 0; s >>= 1) {
        if (tid < s) redA[tid] = fmaxf(redA[tid], redA[tid + s]);
        __syncthreads();
    }
    if (tid == 0) sm = redA[0];
    __syncthreads();
    const float m = sm;

    // pass 2
    float s1 = 0.f, s2 = 0.f, g1 = 0.f, g2 = 0.f;
    for (int i = tid; i < size; i += 256) {
        const int fi = flat_idx[off + i];
        const float x = logits[fi] - m;
        const float e = expf(x);
        s1 += e;  s2 += e * x;
        const float g = good_mask[fi];
        g1 += g * x;  g2 += g;
    }
    redA[tid] = s1; redB[tid] = s2;
    __syncthreads();
    for (int s = 128; s > 0; s >>= 1) {
        if (tid < s) { redA[tid] += redA[tid + s]; redB[tid] += redB[tid + s]; }
        __syncthreads();
    }
    if (tid == 0) { sS1 = redA[0]; sS2 = redB[0]; }
    __syncthreads();
    redA[tid] = g1; redB[tid] = g2;
    __syncthreads();
    for (int s = 128; s > 0; s >>= 1) {
        if (tid < s) { redA[tid] += redA[tid + s]; redB[tid] += redB[tid + s]; }
        __syncthreads();
    }

    if (tid == 0) {
        const float S1 = sS1, S2 = sS2;
        const float G1 = redA[0], G2 = redB[0];
        const float lse = logf(S1);
        if (G2 > 0.f) {
            atomicAdd(&out[0], -(G1 - lse * G2) / fmaxf(G2, 1.f));
            atomicAdd(&out[1], 1.f);
        }
        const int p = sel_flat_pos[b];
        const float lsm_sel = logits[flat_idx[p]] - m - lse;
        const float et = event_time[b];
        atomicAdd(&out[2], TPM_C * et * lsm_sel);
        atomicAdd(&out[3], et);
        const float minus_ent = (S2 / S1 - lse) / logf((float)size);
        atomicAdd(&out[4], EC_C * minus_ent);
    }
}

// ---------------------------------------------------------------------------
// Host launcher
// ---------------------------------------------------------------------------
extern "C" void kernel_launch(void* const* d_in, const int* in_sizes, int n_in,
                              void* d_out, int out_size, void* d_ws, size_t ws_size,
                              hipStream_t stream) {
    const float* features   = (const float*)d_in[0];
    const float* tweak      = (const float*)d_in[1];
    const float* mulW       = (const float*)d_in[2];
    const float* mulb       = (const float*)d_in[3];
    const float* W1         = (const float*)d_in[4];
    const float* b1         = (const float*)d_in[5];
    const float* W2         = (const float*)d_in[6];
    const float* b2         = (const float*)d_in[7];
    const float* key_w      = (const float*)d_in[8];
    const float* good_mask  = (const float*)d_in[9];
    const float* event_time = (const float*)d_in[10];
    const int*   flat_idx   = (const int*)d_in[11];
    /* seg_ids d_in[12] unused: offsets derived from seg_sizes */
    const int*   sel_flat_pos = (const int*)d_in[13];
    const int*   seg_sizes    = (const int*)d_in[14];

    const int N_ = in_sizes[0] / Fdim;
    const int S_ = in_sizes[10];

    // Workspace carve (256B aligned slices)
    char* ws = (char*)d_ws;
    size_t o = 0;
    auto carve = [&](size_t bytes) -> void* {
        o = (o + 255) & ~(size_t)255;
        void* p = ws + o;
        o += bytes;
        return p;
    };
    unsigned short* W1sw = (unsigned short*)carve(sizeof(unsigned short) * Fdim * Hdim);
    float* fm      = (float*)carve(sizeof(float) * Fdim);
    float* fa      = (float*)carve(sizeof(float) * Fdim);
    float* v       = (float*)carve(sizeof(float) * Hdim);
    float* c0      = (float*)carve(sizeof(float));
    float* logits  = (float*)carve(sizeof(float) * (size_t)N_);
    int*   offsets = (int*)carve(sizeof(int) * (size_t)S_);

    float* out = (float*)d_out;

    setup_kernel<<<1, 256, 0, stream>>>(tweak, mulW, mulb, W1, W2, b2, key_w,
                                        W1sw, fm, fa, v, c0, out, S_);
    scan_kernel<<<1, 256, 0, stream>>>(seg_sizes, offsets, S_);
    fused_mlp_logits_kernel<<<(N_ + 31) / 32, 256, 0, stream>>>(
        features, W1sw, b1, fm, fa, v, c0, logits, N_);
    segment_loss_kernel<<<S_, 256, 0, stream>>>(
        logits, flat_idx, good_mask, event_time, sel_flat_pos, seg_sizes,
        offsets, out);
}
